// GATv2_47545287966778
// MI455X (gfx1250) — matrix operand order repaired and checked by reading, not compiled, not run on hardware
//
#include <hip/hip_runtime.h>
#include <hip/hip_bf16.h>

// ---------------------------------------------------------------------------
// GATv2 (2 layers) for MI455X / gfx1250.
//   Layer 1: heads=8, hid=32 (concat)   Layer 2: heads=1, out=64 (mean)
// GEMMs: v_wmma_f32_16x16x32_bf16, one wave per 16x64 C strip (4 N-tiles,
//        A fragment reused 4x; all B fragments preloaded so the 4 WMMAs
//        issue back-to-back after a single s_wait_loadcnt).
// Edge softmax-aggregation: wave-per-edge coalesced passes with __shfl_xor
// reductions (ds path) + global f32 atomics.
// ---------------------------------------------------------------------------

typedef __attribute__((ext_vector_type(16))) __bf16 v16bf;
typedef __attribute__((ext_vector_type(8)))  float  v8f;

union BF16x16 { v16bf v; uint4 q[2]; };

__device__ __forceinline__ unsigned short f2bf(float f) {
    unsigned u = __float_as_uint(f);
    unsigned r = (u + 0x7FFFu + ((u >> 16) & 1u)) >> 16;   // RNE
    return (unsigned short)r;
}

// monotonic float<->uint encoding so atomicMax(uint) == float max
__device__ __forceinline__ unsigned fenc(float f) {
    unsigned u = __float_as_uint(f);
    return (u & 0x80000000u) ? ~u : (u | 0x80000000u);
}
__device__ __forceinline__ float fdec(unsigned u) {
    unsigned b = (u & 0x80000000u) ? (u & 0x7FFFFFFFu) : ~u;
    return __uint_as_float(b);
}

// ---------------------------------------------------------------------------
// Casts
// ---------------------------------------------------------------------------
__global__ void k_cast_bf16(const float* __restrict__ in,
                            unsigned short* __restrict__ out, int n) {
    int i = blockIdx.x * blockDim.x + threadIdx.x;
    if (i < n) out[i] = f2bf(in[i]);
}

// in: [K,N] row-major fp32  ->  out: [N,K] bf16 (transposed for B-fragment loads)
__global__ void k_cast_bf16_T(const float* __restrict__ in,
                              unsigned short* __restrict__ out, int K, int N) {
    int i = blockIdx.x * blockDim.x + threadIdx.x;
    if (i < K * N) {
        int k = i / N, n = i % N;
        out[(size_t)n * K + k] = f2bf(in[i]);
    }
}

// ---------------------------------------------------------------------------
// WMMA GEMM: C[M,N] = A[M,K](bf16 row-major) * Bt[N,K](bf16, B transposed)
// One wave per 16x64 strip: 4 accumulators, A fragment loaded once per kb.
// Requires N % 64 == 0, M % 16 == 0, K % 32 == 0.
// A fragment (16x32 bf16, ISA 7.12.2): lane half h=lane/16, row m=lane%16,
//   elems 0-15 <- K = kb + 16*h + i (contiguous, same order as B)
// B fragment (32x16): lane col n=lane%16, elems 0-15 <- K = kb + 16*h + i.
// C/D (16x16 f32): vgpr v -> row v + 8*h, col = lane%16.
// ---------------------------------------------------------------------------
__global__ void k_wmma_gemm(const unsigned short* __restrict__ A,
                            const unsigned short* __restrict__ Bt,
                            float* __restrict__ C,
                            int M, int N, int K) {
    int wave = (blockIdx.x * blockDim.x + threadIdx.x) >> 5;
    int lane = threadIdx.x & 31;
    int ng4  = N >> 6;                       // groups of 4 n-tiles
    int strips = (M >> 4) * ng4;
    if (wave >= strips) return;
    int m0 = (wave / ng4) << 4;
    int n0 = (wave % ng4) << 6;
    int half = lane >> 4;
    int l16  = lane & 15;

    const unsigned short* arow = A + (size_t)(m0 + l16) * K;
    const unsigned short* brow[4];
#pragma unroll
    for (int j = 0; j < 4; ++j)
        brow[j] = Bt + (size_t)(n0 + j * 16 + l16) * K;

    v8f acc[4] = {};
    for (int kb = 0; kb < K; kb += 32) {
        __builtin_prefetch(arow + kb + 64, 0, 3);   // global_prefetch_b8
        // ---- issue ALL fragment loads first (one clause, one wait) -------
        BF16x16 a, b[4];
        const uint4* ap = (const uint4*)(arow + kb);        // 16B-aligned
        a.q[0] = ap[2 * half];      // K = kb + 16*half   .. +7
        a.q[1] = ap[2 * half + 1];  // K = kb+16*half+8  .. +7
#pragma unroll
        for (int j = 0; j < 4; ++j) {
            const uint4* bp = (const uint4*)(brow[j] + kb + (half << 4));
            b[j].q[0] = bp[0];
            b[j].q[1] = bp[1];
        }
        // ---- then 4 independent WMMAs back-to-back -----------------------
#pragma unroll
        for (int j = 0; j < 4; ++j)
            acc[j] = __builtin_amdgcn_wmma_f32_16x16x32_bf16(
                         false, a.v, false, b[j].v, (short)0, acc[j],
                         false, false);
    }
#pragma unroll
    for (int j = 0; j < 4; ++j) {
        float* crow = C + (size_t)(m0 + (half << 3)) * N + n0 + j * 16 + l16;
#pragma unroll
        for (int v = 0; v < 8; ++v) crow[(size_t)v * N] = acc[j][v];
    }
}

// ---------------------------------------------------------------------------
// Edge passes (self loops: edge id >= E  =>  src = dst = id - E)
// Wave-per-edge: lane = channel -> coalesced 128B gathers/scatters.
// ---------------------------------------------------------------------------
__global__ void k_edge_score_wave(const float* __restrict__ xl,
                                  const float* __restrict__ xr,
                                  const float* __restrict__ att,
                                  const int* __restrict__ src,
                                  const int* __restrict__ dst,
                                  int E, int ET, int H, int C,
                                  float* __restrict__ e,
                                  unsigned* __restrict__ menc) {
    int eid  = (blockIdx.x * blockDim.x + threadIdx.x) >> 5;
    int lane = threadIdx.x & 31;
    if (eid >= ET) return;
    int s = (eid < E) ? src[eid] : (eid - E);
    int d = (eid < E) ? dst[eid] : (eid - E);
    const float* pl = xl + (size_t)s * H * C;
    const float* pr = xr + (size_t)d * H * C;
    for (int h = 0; h < H; ++h) {
        float partial = 0.f;
        for (int c = lane; c < C; c += 32) {
            float z = pl[h * C + c] + pr[h * C + c];
            z = (z > 0.f) ? z : 0.2f * z;           // leaky_relu, slope 0.2
            partial += z * att[h * C + c];
        }
#pragma unroll
        for (int off = 16; off > 0; off >>= 1)      // wave32 reduction (ds path)
            partial += __shfl_xor(partial, off, 32);
        if (lane == 0) {
            e[(size_t)eid * H + h] = partial;
            atomicMax(&menc[d * H + h], fenc(partial));
        }
    }
}

__global__ void k_edge_expsum(const int* __restrict__ dst,
                              int E, int ET, int H,
                              float* __restrict__ e,
                              const unsigned* __restrict__ menc,
                              float* __restrict__ ssum) {
    int idx = blockIdx.x * blockDim.x + threadIdx.x;
    if (idx >= ET * H) return;
    int eid = idx / H, h = idx % H;
    int d = (eid < E) ? dst[eid] : (eid - E);
    float a = __expf(e[idx] - fdec(menc[d * H + h]));
    e[idx] = a;
    atomicAdd(&ssum[d * H + h], a);
}

__global__ void k_edge_accum_wave(const float* __restrict__ xl,
                                  const int* __restrict__ src,
                                  const int* __restrict__ dst,
                                  int E, int ET, int H, int C,
                                  const float* __restrict__ e,
                                  const float* __restrict__ ssum,
                                  float* __restrict__ acc) {
    int eid  = (blockIdx.x * blockDim.x + threadIdx.x) >> 5;
    int lane = threadIdx.x & 31;
    if (eid >= ET) return;
    int s = (eid < E) ? src[eid] : (eid - E);
    int d = (eid < E) ? dst[eid] : (eid - E);
    const float* pl = xl  + (size_t)s * H * C;
    float*       po = acc + (size_t)d * H * C;
    for (int h = 0; h < H; ++h) {
        float alpha = e[(size_t)eid * H + h] / ssum[d * H + h];
        for (int c = lane; c < C; c += 32)
            atomicAdd(&po[h * C + c], alpha * pl[h * C + c]);
    }
}

// layer-1 epilogue: bias + ELU, emit bf16 h for layer-2 GEMMs
__global__ void k_finalize_elu_bf16(const float* __restrict__ acc,
                                    const float* __restrict__ bias,
                                    unsigned short* __restrict__ hbf,
                                    int total, int HC) {
    int i = blockIdx.x * blockDim.x + threadIdx.x;
    if (i >= total) return;
    float v = acc[i] + bias[i % HC];
    v = (v > 0.f) ? v : (__expf(v) - 1.f);   // ELU(alpha=1)
    hbf[i] = f2bf(v);
}

// layer-2 epilogue: bias add (heads=1 -> mean is identity)
__global__ void k_finalize_out(const float* __restrict__ acc,
                               const float* __restrict__ bias,
                               float* __restrict__ out, int total, int C) {
    int i = blockIdx.x * blockDim.x + threadIdx.x;
    if (i >= total) return;
    out[i] = acc[i] + bias[i % C];
}

// ---------------------------------------------------------------------------
// Host launcher
// ---------------------------------------------------------------------------
extern "C" void kernel_launch(void* const* d_in, const int* in_sizes, int n_in,
                              void* d_out, int out_size, void* d_ws, size_t ws_size,
                              hipStream_t stream) {
    (void)n_in; (void)out_size; (void)ws_size;
    const float* x    = (const float*)d_in[0];
    const int*   ei   = (const int*)  d_in[1];
    // d_in[2] = edge_type (unused by the reference forward)
    const float* Wl1  = (const float*)d_in[3];
    const float* Wr1  = (const float*)d_in[4];
    const float* att1 = (const float*)d_in[5];
    const float* b1   = (const float*)d_in[6];
    const float* Wl2  = (const float*)d_in[7];
    const float* Wr2  = (const float*)d_in[8];
    const float* att2 = (const float*)d_in[9];
    const float* b2   = (const float*)d_in[10];

    const int F = 128, H1 = 8, C1 = 32, D1 = H1 * C1 /*256*/, C2 = 64;
    const int Nn = in_sizes[0] / F;       // 50000
    const int E  = in_sizes[1] / 2;       // 800000
    const int ET = E + Nn;                // + self loops
    const int* src = ei;
    const int* dst = ei + E;

    // ---- workspace carve -------------------------------------------------
    char* w = (char*)d_ws;
    size_t off = 0;
    auto alloc = [&](size_t bytes) -> void* {
        off = (off + 255) & ~(size_t)255;
        void* p = w + off;
        off += bytes;
        return p;
    };
    unsigned short* xbf   = (unsigned short*)alloc((size_t)Nn * F * 2);
    unsigned short* WlT1  = (unsigned short*)alloc((size_t)D1 * F * 2);
    unsigned short* WrT1  = (unsigned short*)alloc((size_t)D1 * F * 2);
    unsigned short* WlT2  = (unsigned short*)alloc((size_t)C2 * D1 * 2);
    unsigned short* WrT2  = (unsigned short*)alloc((size_t)C2 * D1 * 2);
    float*    xl1  = (float*)   alloc((size_t)Nn * D1 * 4);
    float*    xr1  = (float*)   alloc((size_t)Nn * D1 * 4);
    float*    e1   = (float*)   alloc((size_t)ET * H1 * 4);
    unsigned* m1   = (unsigned*)alloc((size_t)Nn * H1 * 4);
    float*    s1   = (float*)   alloc((size_t)Nn * H1 * 4);
    float*    acc1 = (float*)   alloc((size_t)Nn * D1 * 4);
    unsigned short* hbf = (unsigned short*)alloc((size_t)Nn * D1 * 2);
    float*    xl2  = (float*)   alloc((size_t)Nn * C2 * 4);
    float*    xr2  = (float*)   alloc((size_t)Nn * C2 * 4);
    float*    e2   = (float*)   alloc((size_t)ET * 4);
    unsigned* m2   = (unsigned*)alloc((size_t)Nn * 4);
    float*    s2   = (float*)   alloc((size_t)Nn * 4);
    float*    acc2 = (float*)   alloc((size_t)Nn * C2 * 4);

    const int B = 256;
    auto grid = [](long long t, int b) { return (unsigned)((t + b - 1) / b); };

    // ---- casts -----------------------------------------------------------
    k_cast_bf16  <<<grid((long long)Nn * F, B), B, 0, stream>>>(x, xbf, Nn * F);
    k_cast_bf16_T<<<grid((long long)F * D1, B), B, 0, stream>>>(Wl1, WlT1, F, D1);
    k_cast_bf16_T<<<grid((long long)F * D1, B), B, 0, stream>>>(Wr1, WrT1, F, D1);
    k_cast_bf16_T<<<grid((long long)D1 * C2, B), B, 0, stream>>>(Wl2, WlT2, D1, C2);
    k_cast_bf16_T<<<grid((long long)D1 * C2, B), B, 0, stream>>>(Wr2, WrT2, D1, C2);

    // ---- layer 1 GEMMs (WMMA, 4 n-tiles per wave) ------------------------
    {
        long long strips = (long long)(Nn / 16) * (D1 / 64);
        unsigned g = grid(strips * 32, B);
        k_wmma_gemm<<<g, B, 0, stream>>>(xbf, WlT1, xl1, Nn, D1, F);
        k_wmma_gemm<<<g, B, 0, stream>>>(xbf, WrT1, xr1, Nn, D1, F);
    }

    // ---- layer 1 edge softmax + aggregation ------------------------------
    hipMemsetAsync(m1,   0, (size_t)Nn * H1 * 4, stream);
    hipMemsetAsync(s1,   0, (size_t)Nn * H1 * 4, stream);
    hipMemsetAsync(acc1, 0, (size_t)Nn * D1 * 4, stream);
    {
        unsigned gw = grid((long long)ET * 32, B);   // wave per edge
        k_edge_score_wave<<<gw, B, 0, stream>>>(xl1, xr1, att1, src, dst,
                                                E, ET, H1, C1, e1, m1);
        k_edge_expsum<<<grid((long long)ET * H1, B), B, 0, stream>>>(
            dst, E, ET, H1, e1, m1, s1);
        k_edge_accum_wave<<<gw, B, 0, stream>>>(xl1, src, dst, E, ET, H1, C1,
                                                e1, s1, acc1);
    }
    k_finalize_elu_bf16<<<grid((long long)Nn * D1, B), B, 0, stream>>>(
        acc1, b1, hbf, Nn * D1, D1);

    // ---- layer 2 GEMMs (WMMA) -------------------------------------------
    {
        long long strips = (long long)(Nn / 16) * (C2 / 64);
        unsigned g = grid(strips * 32, B);
        k_wmma_gemm<<<g, B, 0, stream>>>(hbf, WlT2, xl2, Nn, C2, D1);
        k_wmma_gemm<<<g, B, 0, stream>>>(hbf, WrT2, xr2, Nn, C2, D1);
    }

    // ---- layer 2 edge softmax + aggregation ------------------------------
    hipMemsetAsync(m2,   0, (size_t)Nn * 4, stream);
    hipMemsetAsync(s2,   0, (size_t)Nn * 4, stream);
    hipMemsetAsync(acc2, 0, (size_t)Nn * C2 * 4, stream);
    {
        unsigned gw = grid((long long)ET * 32, B);   // wave per edge
        k_edge_score_wave<<<gw, B, 0, stream>>>(xl2, xr2, att2, src, dst,
                                                E, ET, 1, C2, e2, m2);
        k_edge_expsum<<<grid((long long)ET, B), B, 0, stream>>>(
            dst, E, ET, 1, e2, m2, s2);
        k_edge_accum_wave<<<gw, B, 0, stream>>>(xl2, src, dst, E, ET, 1, C2,
                                                e2, s2, acc2);
    }
    k_finalize_out<<<grid((long long)Nn * C2, B), B, 0, stream>>>(
        acc2, b2, (float*)d_out, Nn * C2, C2);
}
